// BiMambaLayer_12343736008804
// MI455X (gfx1250) — matrix operand order repaired and checked by reading, not compiled, not run on hardware
//
#include <hip/hip_runtime.h>
#include <hip/hip_bf16.h>
#include <math.h>
#include <stdint.h>

// ---------------- problem constants (from reference) ----------------
#define D_MODEL 1024
#define D_INNER 2048
#define D_STATE 16
#define D_CONV  4
#define DT_RANK 64
#define BATCH   2
#define SEQ     2048
#define NROWS   (BATCH * SEQ)   // 4096 activation rows

// ---------------- WMMA types (CDNA5 / gfx1250, wave32) ----------------
typedef __bf16 bf16;
typedef __attribute__((ext_vector_type(16))) __bf16 v16bf;
typedef __attribute__((ext_vector_type(8)))  float  v8f;

__device__ __forceinline__ float softplusf(float v) {
    return (v > 20.f) ? v : log1pf(__expf(v));
}
__device__ __forceinline__ float sigmoidf_(float v) {
    return 1.f / (1.f + __expf(-v));
}

// epilogue / flip flags
#define EP_BIAS     1
#define EP_SOFTPLUS 2
#define EP_ACC      4
#define EP_MIRROR   8      // also store bf16 copy of C (for GEMM->GEMM chaining)
#define FLIP_A      1
#define FLIP_OUT    2

// one 16-byte async global->LDS copy (ASYNCcnt-tracked, bypasses VGPRs)
__device__ __forceinline__ void async_cp16(unsigned lds_off, const bf16* g) {
    asm volatile("global_load_async_to_lds_b128 %0, %1, off"
                 :: "v"(lds_off), "v"((unsigned long long)(uintptr_t)g)
                 : "memory");
}

// =====================================================================
// bf16-WMMA GEMM:  C[M,N] (+)= epilogue( A[M,K] @ Wt[N,K]^T )
//   A : bf16 row-major [M x K]   (lda elements)
//   Wt: bf16, weights pre-transposed to [N x K] (ldw elements), N padded
//       to a multiple of 128 with zero rows so tile copies never go OOB.
// Block tile 128x128, K-step 32, double-buffered LDS fed by
// GLOBAL_LOAD_ASYNC_TO_LDS_B128; 8 waves x (2x4) 16x16 WMMA frags.
// =====================================================================
__launch_bounds__(256)
__global__ void gemm_wmma_kernel(const bf16* __restrict__ A, int lda,
                                 const bf16* __restrict__ Wt, int ldw,
                                 const float* __restrict__ bias,
                                 float* __restrict__ C, int ldc,
                                 bf16* __restrict__ Cbf, int ldcb,
                                 int M, int N, int K, int ep, int flip)
{
    __shared__ bf16 sA[2][128 * 32];   // [m][k] tiles, 8KB each
    __shared__ bf16 sB[2][128 * 32];   // [n][k] tiles, 8KB each

    const int tid  = threadIdx.x;
    const int lane = tid & 31;
    const int wave = tid >> 5;
    const int wm   = wave & 3;
    const int wn   = wave >> 2;
    const int mtile = blockIdx.y * 128;
    const int ntile = blockIdx.x * 128;
    const int l16  = lane & 15;
    const int half = lane >> 4;

    // staging: 512 chunks of 16B per tile; this thread owns chunks c0,c1.
    // chunk c -> row c>>2, 16B piece (c&3); LDS element offset = c*8.
    const int c0 = tid, c1 = tid + 256;
    int am0 = mtile + (c0 >> 2), am1 = mtile + (c1 >> 2);
    if (flip & FLIP_A) {
        int b0 = am0 / SEQ, t0 = am0 % SEQ; am0 = b0 * SEQ + (SEQ - 1 - t0);
        int b1 = am1 / SEQ, t1 = am1 % SEQ; am1 = b1 * SEQ + (SEQ - 1 - t1);
    }
    const int bn0 = ntile + (c0 >> 2), bn1 = ntile + (c1 >> 2);

    auto issue = [&](int buf, int k0) {
        async_cp16((unsigned)(uintptr_t)&sA[buf][c0 * 8],
                   A + (long)am0 * lda + k0 + (c0 & 3) * 8);
        async_cp16((unsigned)(uintptr_t)&sA[buf][c1 * 8],
                   A + (long)am1 * lda + k0 + (c1 & 3) * 8);
        async_cp16((unsigned)(uintptr_t)&sB[buf][c0 * 8],
                   Wt + (long)bn0 * ldw + k0 + (c0 & 3) * 8);
        async_cp16((unsigned)(uintptr_t)&sB[buf][c1 * 8],
                   Wt + (long)bn1 * ldw + k0 + (c1 & 3) * 8);
    };

    v8f acc[2][4];
    #pragma unroll
    for (int i = 0; i < 2; ++i)
        #pragma unroll
        for (int j = 0; j < 4; ++j) acc[i][j] = v8f{};

    int cur = 0;
    issue(0, 0);
    for (int k0 = 0; k0 < K; k0 += 32) {
        const bool more = (k0 + 32) < K;
        if (more) {
            issue(cur ^ 1, k0 + 32);                 // prefetch next tile
            asm volatile("s_wait_asynccnt 4" ::: "memory"); // cur's 4 done
        } else {
            asm volatile("s_wait_asynccnt 0" ::: "memory");
        }
        __syncthreads();

        // ---- A fragments (ISA 16-bit A 16x32 layout; pairs -> b32) ----
        v16bf af[2];
        #pragma unroll
        for (int fm = 0; fm < 2; ++fm) {
            const unsigned* rowp =
                (const unsigned*)&sA[cur][(wm * 32 + fm * 16 + l16) * 32];
            #pragma unroll
            for (int v = 0; v < 8; ++v) {
                int kp = ((v < 4) ? 0 : 8) + half * 4 + (v & 3);
                unsigned pr = rowp[kp];
                af[fm][2 * v]     = __builtin_bit_cast(__bf16, (unsigned short)(pr & 0xffffu));
                af[fm][2 * v + 1] = __builtin_bit_cast(__bf16, (unsigned short)(pr >> 16));
            }
        }
        // ---- B fragments (same k order as A: K=8h..8h+7, 16+8h..16+8h+7) ----
        v16bf bfrag[4];
        #pragma unroll
        for (int fn = 0; fn < 4; ++fn) {
            const unsigned* colp =
                (const unsigned*)&sB[cur][(wn * 64 + fn * 16 + l16) * 32];
            #pragma unroll
            for (int v = 0; v < 8; ++v) {
                unsigned pr = colp[((v < 4) ? 0 : 8) + half * 4 + (v & 3)];
                bfrag[fn][2 * v]     = __builtin_bit_cast(__bf16, (unsigned short)(pr & 0xffffu));
                bfrag[fn][2 * v + 1] = __builtin_bit_cast(__bf16, (unsigned short)(pr >> 16));
            }
        }
        #pragma unroll
        for (int fm = 0; fm < 2; ++fm)
            #pragma unroll
            for (int fn = 0; fn < 4; ++fn)
                acc[fm][fn] = __builtin_amdgcn_wmma_f32_16x16x32_bf16(
                    false, af[fm], false, bfrag[fn],
                    (short)0, acc[fm][fn], false, false);

        __syncthreads();   // everyone done with cur before it is re-targeted
        cur ^= 1;
    }

    // ---- epilogue: VGPR r -> row r (lanes<16) / r+8 (lanes>=16) ----
    #pragma unroll
    for (int fm = 0; fm < 2; ++fm) {
        #pragma unroll
        for (int fn = 0; fn < 4; ++fn) {
            int col = ntile + wn * 64 + fn * 16 + l16;
            #pragma unroll
            for (int r = 0; r < 8; ++r) {
                int lrow = mtile + wm * 32 + fm * 16 + r + 8 * half;
                if (lrow < M && col < N) {
                    float v = acc[fm][fn][r];
                    if (ep & EP_BIAS)     v += bias[col];
                    if (ep & EP_SOFTPLUS) v = softplusf(v);
                    int srow = lrow;
                    if (flip & FLIP_OUT) { int b = lrow / SEQ, t = lrow % SEQ; srow = b * SEQ + (SEQ - 1 - t); }
                    long o = (long)srow * ldc + col;
                    if (ep & EP_ACC) C[o] += v;
                    else             C[o]  = v;
                    if (ep & EP_MIRROR) Cbf[(long)srow * ldcb + col] = (bf16)v;
                }
            }
        }
    }
}

// ============ fp32 -> bf16 convert (activations) ============
__global__ void cvt_bf16_kernel(const float* __restrict__ src,
                                bf16* __restrict__ dst, int total)
{
    int i = blockIdx.x * blockDim.x + threadIdx.x;
    if (i < total) dst[i] = (bf16)src[i];
}

// ===== weight transpose+convert: src[K][N] fp32 -> dst[Npad][K] bf16 =====
__global__ void wt_transpose_kernel(const float* __restrict__ src,
                                    bf16* __restrict__ dst,
                                    int K, int N, int total /* = Npad*K */)
{
    int i = blockIdx.x * blockDim.x + threadIdx.x;
    if (i >= total) return;
    int n = i / K, k = i % K;
    dst[i] = (n < N) ? (bf16)src[(long)k * N + n] : (bf16)0.f;
}

// ===== quality gate: xg = x * d/(1+d), d = softplus(dlin)*exp(-a*u) =====
// output written directly as bf16 (only consumed by WMMA GEMM)
__global__ void gate_kernel(const float* __restrict__ x,
                            const float* __restrict__ u,
                            const float* __restrict__ alpha_p,
                            const float* __restrict__ dlin,
                            bf16* __restrict__ xgbf, int total)
{
    int i = blockIdx.x * blockDim.x + threadIdx.x;
    if (i >= total) return;
    int r = i / D_MODEL;
    float a  = alpha_p[0];
    float dr = softplusf(dlin[i]);
    float di = dr * __expf(-a * u[r]);
    xgbf[i] = (bf16)(x[i] * (di / (1.f + di)));
}

// ===== causal depthwise conv1d (w=4) + SiLU; dual fp32/bf16 output =====
__global__ void conv_silu_kernel(const float* __restrict__ xi,   // ld 2*D_INNER
                                 const float* __restrict__ cw,
                                 const float* __restrict__ cb,
                                 float* __restrict__ xc,
                                 bf16* __restrict__ xcbf, int total)
{
    int i = blockIdx.x * blockDim.x + threadIdx.x;
    if (i >= total) return;
    int r = i / D_INNER, c = i % D_INNER;
    int b = r / SEQ, t = r % SEQ;
    float acc = cb[c];
    #pragma unroll
    for (int j = 0; j < D_CONV; ++j) {
        int tt = t - (D_CONV - 1) + j;
        if (tt >= 0)
            acc += xi[(long)(b * SEQ + tt) * (2 * D_INNER) + c] * cw[c * D_CONV + j];
    }
    float v = acc * sigmoidf_(acc);
    xc[i]   = v;
    xcbf[i] = (bf16)v;
}

// ===== selective scan; ys written as bf16 (only consumed by GEMM) =====
__launch_bounds__(256)
__global__ void scan_kernel(const float* __restrict__ dt,
                            const float* __restrict__ xc,
                            const float* __restrict__ xdb,   // [.,96]: dt_lo|B|C
                            const float* __restrict__ z,     // ld 2*D_INNER
                            const float* __restrict__ A_log,
                            const float* __restrict__ Dp,
                            bf16* __restrict__ ysbf)
{
    int gid = blockIdx.x * blockDim.x + threadIdx.x;   // 0 .. B*D_INNER-1
    int b = gid / D_INNER, ch = gid % D_INNER;
    float As[D_STATE], h[D_STATE];
    #pragma unroll
    for (int s = 0; s < D_STATE; ++s) {
        As[s] = -__expf(A_log[ch * D_STATE + s]);
        h[s]  = 0.f;
    }
    float dpc = Dp[ch];
    for (int t = 0; t < SEQ; ++t) {
        long row = (long)b * SEQ + t;
        float dtv = dt[row * D_INNER + ch];
        float xv  = xc[row * D_INNER + ch];
        float zv  = z [row * (2 * D_INNER) + ch];
        const float* bc = &xdb[row * 96];
        float y = 0.f;
        #pragma unroll
        for (int s = 0; s < D_STATE; ++s) {
            float bs = bc[DT_RANK + s];
            float cs = bc[DT_RANK + D_STATE + s];
            h[s] = __expf(dtv * As[s]) * h[s] + dtv * bs * xv;
            y += h[s] * cs;
        }
        float yv = (y + xv * dpc) * (zv * sigmoidf_(zv));
        ysbf[row * D_INNER + ch] = (bf16)yv;
    }
}

// =====================================================================
// host-side orchestration
// =====================================================================
extern "C" void kernel_launch(void* const* d_in, const int* in_sizes, int n_in,
                              void* d_out, int out_size, void* d_ws, size_t ws_size,
                              hipStream_t stream)
{
    (void)in_sizes; (void)n_in; (void)out_size; (void)ws_size;
    const float* x       = (const float*)d_in[0];
    const float* u       = (const float*)d_in[1];
    const float* alpha   = (const float*)d_in[2];
    const float* W_delta = (const float*)d_in[3];
    const float* b_delta = (const float*)d_in[4];
    const float* W_proj  = (const float*)d_in[5];
    const float* b_proj  = (const float*)d_in[6];

    float* out     = (float*)d_out;
    float* fwd_out = out + (size_t)NROWS * D_MODEL;
    float* bwd_out = out + 2 * (size_t)NROWS * D_MODEL;

    // ---- fp32 scratch ----
    float* wsf = (float*)d_ws;
    size_t fo = 0;
    float* xz  = wsf + fo; fo += (size_t)NROWS * 2 * D_INNER;  // [xi | z]
    float* xc  = wsf + fo; fo += (size_t)NROWS * D_INNER;
    float* xdb = wsf + fo; fo += (size_t)NROWS * 96;
    float* dt  = wsf + fo; fo += (size_t)NROWS * D_INNER;
    float* dlin = dt;  // alias: fully consumed by gate before dt is written

    // ---- bf16 scratch ----
    bf16* wsb = (bf16*)(wsf + fo);
    size_t bo = 0;
    bf16* xbf    = wsb + bo; bo += (size_t)NROWS * D_MODEL;
    bf16* xgbf   = wsb + bo; bo += (size_t)NROWS * D_MODEL;
    bf16* xcbf   = wsb + bo; bo += (size_t)NROWS * D_INNER;
    bf16* xdbbf  = wsb + bo; bo += (size_t)NROWS * 96;
    bf16* ysbf   = wsb + bo; bo += (size_t)NROWS * D_INNER;
    bf16* fwdbf  = wsb + bo; bo += (size_t)NROWS * D_MODEL;
    bf16* bwdbf  = wsb + bo; bo += (size_t)NROWS * D_MODEL;
    bf16* wt_dl  = wsb + bo; bo += (size_t)D_MODEL * D_MODEL;      // W_delta^T
    bf16* wt_in  = wsb + bo; bo += (size_t)(2 * D_INNER) * D_MODEL;
    bf16* wt_x   = wsb + bo; bo += (size_t)128 * D_INNER;          // N=96 -> 128
    bf16* wt_dt  = wsb + bo; bo += (size_t)D_INNER * DT_RANK;
    bf16* wt_out = wsb + bo; bo += (size_t)D_MODEL * D_INNER;
    bf16* wt_pj  = wsb + bo; bo += (size_t)D_MODEL * D_MODEL;      // reused x2

    dim3 blk(256);
    auto gg = [](int M, int N) { return dim3((unsigned)((N + 127) / 128),
                                             (unsigned)((M + 127) / 128)); };
    auto ew = [](size_t n) { return dim3((unsigned)((n + 255) / 256)); };

    // 0) bf16 copies of x and W_delta^T
    cvt_bf16_kernel<<<ew((size_t)NROWS * D_MODEL), blk, 0, stream>>>(
        x, xbf, NROWS * D_MODEL);
    wt_transpose_kernel<<<ew((size_t)D_MODEL * D_MODEL), blk, 0, stream>>>(
        W_delta, wt_dl, D_MODEL, D_MODEL, D_MODEL * D_MODEL);

    // 1) dlin = x @ W_delta + b_delta ; gate -> xgbf
    gemm_wmma_kernel<<<gg(NROWS, D_MODEL), blk, 0, stream>>>(
        xbf, D_MODEL, wt_dl, D_MODEL, b_delta, dlin, D_MODEL,
        nullptr, 0, NROWS, D_MODEL, D_MODEL, EP_BIAS, 0);
    gate_kernel<<<ew((size_t)NROWS * D_MODEL), blk, 0, stream>>>(
        x, u, alpha, dlin, xgbf, NROWS * D_MODEL);

    // 2) two mamba passes; bwd folds the time flip into GEMM addressing
    for (int dir = 0; dir < 2; ++dir) {
        const float* W_in   = (const float*)d_in[7 + 9 * dir + 0];
        const float* conv_w = (const float*)d_in[7 + 9 * dir + 1];
        const float* conv_b = (const float*)d_in[7 + 9 * dir + 2];
        const float* W_x    = (const float*)d_in[7 + 9 * dir + 3];
        const float* W_dt   = (const float*)d_in[7 + 9 * dir + 4];
        const float* b_dt   = (const float*)d_in[7 + 9 * dir + 5];
        const float* A_log  = (const float*)d_in[7 + 9 * dir + 6];
        const float* Dp     = (const float*)d_in[7 + 9 * dir + 7];
        const float* W_out  = (const float*)d_in[7 + 9 * dir + 8];
        int fa = dir ? FLIP_A : 0;
        int fo2 = dir ? FLIP_OUT : 0;
        float* dirout = dir ? bwd_out : fwd_out;
        bf16*  dirbf  = dir ? bwdbf   : fwdbf;

        wt_transpose_kernel<<<ew((size_t)(2 * D_INNER) * D_MODEL), blk, 0, stream>>>(
            W_in, wt_in, D_MODEL, 2 * D_INNER, (2 * D_INNER) * D_MODEL);
        gemm_wmma_kernel<<<gg(NROWS, 2 * D_INNER), blk, 0, stream>>>(
            xgbf, D_MODEL, wt_in, D_MODEL, nullptr, xz, 2 * D_INNER,
            nullptr, 0, NROWS, 2 * D_INNER, D_MODEL, 0, fa);

        conv_silu_kernel<<<ew((size_t)NROWS * D_INNER), blk, 0, stream>>>(
            xz, conv_w, conv_b, xc, xcbf, NROWS * D_INNER);

        wt_transpose_kernel<<<ew((size_t)128 * D_INNER), blk, 0, stream>>>(
            W_x, wt_x, D_INNER, 96, 128 * D_INNER);
        gemm_wmma_kernel<<<gg(NROWS, 96), blk, 0, stream>>>(
            xcbf, D_INNER, wt_x, D_INNER, nullptr, xdb, 96,
            xdbbf, 96, NROWS, 96, D_INNER, EP_MIRROR, 0);

        wt_transpose_kernel<<<ew((size_t)D_INNER * DT_RANK), blk, 0, stream>>>(
            W_dt, wt_dt, DT_RANK, D_INNER, D_INNER * DT_RANK);
        gemm_wmma_kernel<<<gg(NROWS, D_INNER), blk, 0, stream>>>(
            xdbbf, 96, wt_dt, DT_RANK, b_dt, dt, D_INNER,
            nullptr, 0, NROWS, D_INNER, DT_RANK, EP_BIAS | EP_SOFTPLUS, 0);

        scan_kernel<<<ew((size_t)BATCH * D_INNER), blk, 0, stream>>>(
            dt, xc, xdb, xz + D_INNER, A_log, Dp, ysbf);

        wt_transpose_kernel<<<ew((size_t)D_MODEL * D_INNER), blk, 0, stream>>>(
            W_out, wt_out, D_INNER, D_MODEL, D_MODEL * D_INNER);
        gemm_wmma_kernel<<<gg(NROWS, D_MODEL), blk, 0, stream>>>(
            ysbf, D_INNER, wt_out, D_INNER, nullptr, dirout, D_MODEL,
            dirbf, D_MODEL, NROWS, D_MODEL, D_INNER, EP_MIRROR, fo2);
    }

    // 3) out = [fwd|bwd] @ W_proj + b_proj  (split over K halves)
    wt_transpose_kernel<<<ew((size_t)D_MODEL * D_MODEL), blk, 0, stream>>>(
        W_proj, wt_pj, D_MODEL, D_MODEL, D_MODEL * D_MODEL);
    gemm_wmma_kernel<<<gg(NROWS, D_MODEL), blk, 0, stream>>>(
        fwdbf, D_MODEL, wt_pj, D_MODEL, b_proj, out, D_MODEL,
        nullptr, 0, NROWS, D_MODEL, D_MODEL, EP_BIAS, 0);
    wt_transpose_kernel<<<ew((size_t)D_MODEL * D_MODEL), blk, 0, stream>>>(
        W_proj + (size_t)D_MODEL * D_MODEL, wt_pj, D_MODEL, D_MODEL,
        D_MODEL * D_MODEL);
    gemm_wmma_kernel<<<gg(NROWS, D_MODEL), blk, 0, stream>>>(
        bwdbf, D_MODEL, wt_pj, D_MODEL, nullptr, out, D_MODEL,
        nullptr, 0, NROWS, D_MODEL, D_MODEL, EP_ACC, 0);
}
